// SharedSelfAttentionModel_37477884625071
// MI455X (gfx1250) — matrix operand order repaired and checked by reading, not compiled, not run on hardware
//
#include <hip/hip_runtime.h>

// ---------------- WMMA fragment helpers (gfx1250, wave32) ----------------
typedef __attribute__((ext_vector_type(16))) _Float16 v16h;
typedef __attribute__((ext_vector_type(8)))  _Float16 v8h;
typedef __attribute__((ext_vector_type(4)))  _Float16 v4h;
typedef __attribute__((ext_vector_type(8)))  float    v8f;

#define DI __device__ __forceinline__

DI int lane_id() { return (int)(threadIdx.x & 31); }

DI v8f wmma16(v16h a, v16h b, v8f c) {
  return __builtin_amdgcn_wmma_f32_16x16x32_f16(false, a, false, b, (short)0, c, false, false);
}

// A fragment 16x32 from row-major f16: per lane one contiguous 32B run (K = kb..kb+15) -> 2x b128.
DI v16h load_a16(const _Float16* src, int lda) {
  int l = lane_id();
  int m = l & 15;
  int kb = (l < 16) ? 0 : 16;
  const _Float16* row = src + (long)m * lda + kb;
  v8h lo = *(const v8h*)row;          // K = kb .. kb+7
  v8h hi = *(const v8h*)(row + 8);    // K = kb+8 .. kb+15
  v16h a;
#pragma unroll
  for (int j = 0; j < 8; ++j) { a[j] = lo[j]; a[8 + j] = hi[j]; }
  return a;
}

// B^T fragment from row-major f16 (logical B[k][n] = src[n*ld + k]).
// v16h element i <-> K = kb+i: one contiguous 32B run per lane -> 2x b128.
DI v16h load_bt16(const _Float16* src, int ld) {
  int l = lane_id();
  int n = l & 15;
  int kb = (l < 16) ? 0 : 16;
  return *(const v16h*)(src + (long)n * ld + kb);
}

// B fragment 32x16 from KxN row-major f16 (attention V): strided over K, d16 loads.
DI v16h load_b16(const _Float16* src, int ld) {
  int l = lane_id();
  int n = l & 15;
  int kb = (l < 16) ? 0 : 16;
  const _Float16* p = src + (long)kb * ld + n;
  v16h b;
#pragma unroll
  for (int j = 0; j < 8; ++j) {
    b[2 * j]     = p[(long)(2 * j) * ld];
    b[2 * j + 1] = p[(long)(2 * j + 1) * ld];
  }
  return b;
}

// Pre-packed B fragment: lane-contiguous 32B -> 2x b128.
DI v16h load_bp(const _Float16* frag) {
  return *((const v16h*)frag + lane_id());
}

// ---------------- weight pre-pack: row-major f32 (KxN) -> fragment-ordered f16 ----------------
__global__ __launch_bounds__(256)
void k_pack_b(const float* __restrict__ W, _Float16* __restrict__ P,
              int K, int Kpad, int N, int ldb) {
  int KT = Kpad >> 5;
  int wave = (int)((blockIdx.x * blockDim.x + threadIdx.x) >> 5);
  int total = (N >> 4) * KT;
  if (wave >= total) return;
  int n16 = wave / KT, kt = wave % KT;
  int l = lane_id();
  int n = n16 * 16 + (l & 15);
  int kb = kt * 32 + ((l < 16) ? 0 : 16);
  v16h v;
#pragma unroll
  for (int j = 0; j < 8; ++j) {
    int k0 = kb + 2 * j, k1 = k0 + 1;
    float x0 = W[(long)((k0 < K) ? k0 : 0) * ldb + n];
    float x1 = W[(long)((k1 < K) ? k1 : 0) * ldb + n];
    v[2 * j]     = (_Float16)((k0 < K) ? x0 : 0.0f);
    v[2 * j + 1] = (_Float16)((k1 < K) ? x1 : 0.0f);
  }
  *(((v16h*)P) + (long)wave * 32 + l) = v;
}

// ---------------- pad x (B,2,S,20) f32 -> (B,2,S,32) f16 with zeros ----------------
__global__ void k_pad_x(const float* __restrict__ x, _Float16* __restrict__ xp, int total) {
  int i = (int)(blockIdx.x * blockDim.x + threadIdx.x);
  if (i >= total) return;
  int col = i & 31;
  int row = i >> 5;
  xp[i] = (_Float16)((col < 20) ? x[(long)row * 20 + col] : 0.0f);
}

// ---------------- batched GEMM: 64x32 macro-tile per wave, f16 A + packed f16 B ----------------
__global__ __launch_bounds__(256)
void k_gemm(const _Float16* __restrict__ A, const _Float16* __restrict__ Bp,
            const float* __restrict__ bias, float* __restrict__ C,
            _Float16* __restrict__ C16,
            int M, int N, int K, int lda, int ldc, long sA, long sC,
            int relu, int kChunk, int atomicMode) {
  int nt = N >> 5;
  int mt = M >> 6;
  int wave = (int)((blockIdx.x * blockDim.x + threadIdx.x) >> 5);
  if (wave >= mt * nt) return;              // wave-uniform
  int tm = wave / nt, tn = wave % nt;
  int KT = K >> 5;
  const _Float16* Ab = A + (long)blockIdx.y * sA + (long)tm * 64 * lda;
  const _Float16* B0 = Bp + ((long)(tn * 2)     * KT) * 512;
  const _Float16* B1 = Bp + ((long)(tn * 2 + 1) * KT) * 512;
  long cbase = (long)blockIdx.y * sC + (long)tm * 64 * ldc + tn * 32;

  int k0   = (int)blockIdx.z * kChunk;
  int kEnd = min(K, k0 + kChunk);

  v8f c[4][2] = {};
  for (; k0 < kEnd; k0 += 32) {
    long kt = k0 >> 5;
    v16h b0 = load_bp(B0 + kt * 512);
    v16h b1 = load_bp(B1 + kt * 512);
    v16h a0 = load_a16(Ab + k0, lda);
    v16h a1 = load_a16(Ab + (long)16 * lda + k0, lda);
    v16h a2 = load_a16(Ab + (long)32 * lda + k0, lda);
    v16h a3 = load_a16(Ab + (long)48 * lda + k0, lda);
    c[0][0] = wmma16(a0, b0, c[0][0]); c[0][1] = wmma16(a0, b1, c[0][1]);
    c[1][0] = wmma16(a1, b0, c[1][0]); c[1][1] = wmma16(a1, b1, c[1][1]);
    c[2][0] = wmma16(a2, b0, c[2][0]); c[2][1] = wmma16(a2, b1, c[2][1]);
    c[3][0] = wmma16(a3, b0, c[3][0]); c[3][1] = wmma16(a3, b1, c[3][1]);
  }

  int l = lane_id();
  int n = l & 15;
  int mb = (l < 16) ? 0 : 8;
  if (atomicMode) {
    float* Cb = C + cbase;
#pragma unroll
    for (int i = 0; i < 4; ++i) {
#pragma unroll
      for (int r = 0; r < 8; ++r) {
        int m = i * 16 + mb + r;
        unsafeAtomicAdd(&Cb[(long)m * ldc + n],      c[i][0][r]);
        unsafeAtomicAdd(&Cb[(long)m * ldc + 16 + n], c[i][1][r]);
      }
    }
  } else {
    float bv0 = bias ? bias[tn * 32 + n] : 0.0f;
    float bv1 = bias ? bias[tn * 32 + 16 + n] : 0.0f;
#pragma unroll
    for (int i = 0; i < 4; ++i) {
#pragma unroll
      for (int r = 0; r < 8; ++r) {
        int m = i * 16 + mb + r;
        float v0 = c[i][0][r] + bv0;
        float v1 = c[i][1][r] + bv1;
        if (relu) { v0 = fmaxf(v0, 0.0f); v1 = fmaxf(v1, 0.0f); }
        if (C) {
          C[cbase + (long)m * ldc + n]      = v0;
          C[cbase + (long)m * ldc + 16 + n] = v1;
        }
        if (C16) {
          C16[cbase + (long)m * ldc + n]      = (_Float16)v0;
          C16[cbase + (long)m * ldc + 16 + n] = (_Float16)v1;
        }
      }
    }
  }
}

// ---------------- fused ALiBi flash attention (all-f16 operands, f32 softmax) ----------------
// qkv16: (BC, S, 3*512) f16 rows = [q(H,64) | k(H,64) | v(H,64)].
__global__ __launch_bounds__(128)
void k_attn(const _Float16* __restrict__ qkv, float* __restrict__ out, int totalWaves) {
  const int S_ = 512, HD_ = 64, H_ = 8, TRIP = 1536, QT = S_ / 16;
  __shared__ _Float16 pbuf[4][16 * 32];
  int wl = (int)(threadIdx.x >> 5);
  int wg = (int)blockIdx.x * 4 + wl;
  if (wg >= totalWaves) return;
  int qt = wg % QT; int t1 = wg / QT;
  int h = t1 % H_;  int bc = t1 / H_;

  const _Float16* qp = qkv + ((long)bc * S_ + (long)qt * 16) * TRIP + h * HD_;
  const _Float16* kp = qkv + (long)bc * S_ * TRIP + 512  + h * HD_;
  const _Float16* vp = qkv + (long)bc * S_ * TRIP + 1024 + h * HD_;

  v16h aq0 = load_a16(qp,      TRIP);
  v16h aq1 = load_a16(qp + 32, TRIP);

  const float scale = 0.125f;              // 1/sqrt(64)
  float slope = (h < 4) ? 1.0f : 0.5f;     // ALiBi slope 1/2^(h//4)
  int l = lane_id(); int n = l & 15; int mb = (l < 16) ? 0 : 8;
  int q0 = qt * 16;

  v8f o0 = {}, o1 = {}, o2 = {}, o3 = {};
  float mrow[8], lrow[8];
#pragma unroll
  for (int r = 0; r < 8; ++r) { mrow[r] = -1e30f; lrow[r] = 0.0f; }
  _Float16* pw = &pbuf[wl][0];

  for (int kt = 0; kt < 16; ++kt) {
    int kbase = kt * 32;
    v8f s0 = {}, s1 = {};
    const _Float16* k0p = kp + (long)kbase * TRIP;
    s0 = wmma16(aq0, load_bt16(k0p,      TRIP), s0);
    s0 = wmma16(aq1, load_bt16(k0p + 32, TRIP), s0);
    const _Float16* k1p = kp + (long)(kbase + 16) * TRIP;
    s1 = wmma16(aq0, load_bt16(k1p,      TRIP), s1);
    s1 = wmma16(aq1, load_bt16(k1p + 32, TRIP), s1);

    float nm[8];
#pragma unroll
    for (int r = 0; r < 8; ++r) {
      int qrow = q0 + mb + r;
      s0[r] = s0[r] * scale - slope * fabsf((float)(qrow - (kbase + n)));
      s1[r] = s1[r] * scale - slope * fabsf((float)(qrow - (kbase + 16 + n)));
      nm[r] = fmaxf(s0[r], s1[r]);
    }
#pragma unroll
    for (int off = 1; off < 16; off <<= 1) {
#pragma unroll
      for (int r = 0; r < 8; ++r) nm[r] = fmaxf(nm[r], __shfl_xor(nm[r], off, 32));
    }
    float psum[8];
#pragma unroll
    for (int r = 0; r < 8; ++r) {
      float mnew = fmaxf(mrow[r], nm[r]);
      float corr = __expf(mrow[r] - mnew);
      mrow[r] = mnew;
      s0[r] = __expf(s0[r] - mnew);
      s1[r] = __expf(s1[r] - mnew);
      psum[r] = s0[r] + s1[r];
      lrow[r] *= corr;
      o0[r] *= corr; o1[r] *= corr; o2[r] *= corr; o3[r] *= corr;
    }
#pragma unroll
    for (int off = 1; off < 16; off <<= 1) {
#pragma unroll
      for (int r = 0; r < 8; ++r) psum[r] += __shfl_xor(psum[r], off, 32);
    }
#pragma unroll
    for (int r = 0; r < 8; ++r) lrow[r] += psum[r];

    // P (C-layout, f32 regs) -> LDS f16 row-major -> A fragment for P@V
#pragma unroll
    for (int r = 0; r < 8; ++r) {
      pw[(mb + r) * 32 + n]      = (_Float16)s0[r];
      pw[(mb + r) * 32 + 16 + n] = (_Float16)s1[r];
    }
    asm volatile("s_wait_dscnt 0" ::: "memory");
    v16h ap = load_a16(pw, 32);

    const _Float16* vb = vp + (long)kbase * TRIP;
    o0 = wmma16(ap, load_b16(vb,      TRIP), o0);
    o1 = wmma16(ap, load_b16(vb + 16, TRIP), o1);
    o2 = wmma16(ap, load_b16(vb + 32, TRIP), o2);
    o3 = wmma16(ap, load_b16(vb + 48, TRIP), o3);
  }

  float* op = out + ((long)bc * S_ + q0) * 512 + h * HD_;
#pragma unroll
  for (int r = 0; r < 8; ++r) {
    float inv = 1.0f / lrow[r];
    int m = mb + r;
    op[(long)m * 512 + n]      = o0[r] * inv;
    op[(long)m * 512 + 16 + n] = o1[r] * inv;
    op[(long)m * 512 + 32 + n] = o2[r] * inv;
    op[(long)m * 512 + 48 + n] = o3[r] * inv;
  }
}

// ---------------- residual add + LayerNorm: f32 stats, dual f32+f16 output ----------------
__global__ __launch_bounds__(256)
void k_add_ln(const float* __restrict__ res, const float* __restrict__ f,
              const float* __restrict__ g, const float* __restrict__ bb,
              float* __restrict__ out, _Float16* __restrict__ out16, int rows) {
  int wave = (int)((blockIdx.x * blockDim.x + threadIdx.x) >> 5);
  if (wave >= rows) return;
  int l = lane_id();
  const float4* r4 = (const float4*)(res + (long)wave * 512);
  const float4* f4 = (const float4*)(f   + (long)wave * 512);
  float4 v[4];
  float s = 0.0f;
#pragma unroll
  for (int i = 0; i < 4; ++i) {
    float4 a = r4[l + i * 32], b = f4[l + i * 32];
    v[i].x = a.x + b.x; v[i].y = a.y + b.y; v[i].z = a.z + b.z; v[i].w = a.w + b.w;
    s += v[i].x + v[i].y + v[i].z + v[i].w;
  }
#pragma unroll
  for (int off = 1; off < 32; off <<= 1) s += __shfl_xor(s, off, 32);
  float mean = s * (1.0f / 512.0f);
  float vs = 0.0f;
#pragma unroll
  for (int i = 0; i < 4; ++i) {
    float dx = v[i].x - mean, dy = v[i].y - mean, dz = v[i].z - mean, dw = v[i].w - mean;
    vs += dx * dx + dy * dy + dz * dz + dw * dw;
  }
#pragma unroll
  for (int off = 1; off < 32; off <<= 1) vs += __shfl_xor(vs, off, 32);
  float rstd = rsqrtf(vs * (1.0f / 512.0f) + 1e-5f);
  const float4* g4 = (const float4*)g;
  const float4* b4 = (const float4*)bb;
  float4* o4 = (float4*)(out + (long)wave * 512);
  _Float16* o16 = out16 + (long)wave * 512;
#pragma unroll
  for (int i = 0; i < 4; ++i) {
    int c = l + i * 32;
    float4 gg = g4[c], bv = b4[c], o;
    o.x = (v[i].x - mean) * rstd * gg.x + bv.x;
    o.y = (v[i].y - mean) * rstd * gg.y + bv.y;
    o.z = (v[i].z - mean) * rstd * gg.z + bv.z;
    o.w = (v[i].w - mean) * rstd * gg.w + bv.w;
    o4[c] = o;
    v4h hv;
    hv[0] = (_Float16)o.x; hv[1] = (_Float16)o.y;
    hv[2] = (_Float16)o.z; hv[3] = (_Float16)o.w;
    *(v4h*)(o16 + (long)c * 4) = hv;          // packed b64 store
  }
}

// ---------------- row init with bias (for split-K accumulation) ----------------
__global__ void k_row_init(float* __restrict__ C, const float* __restrict__ bias,
                           int total, int cols) {
  int i = (int)(blockIdx.x * blockDim.x + threadIdx.x);
  if (i >= total) return;
  C[i] = bias[i % cols];
}

// ---------------- (optional relu) + batchnorm, optional f16 mirror ----------------
__global__ void k_bn(float* __restrict__ z, _Float16* __restrict__ z16,
                     const float* __restrict__ g, const float* __restrict__ b,
                     const float* __restrict__ m, const float* __restrict__ v,
                     int total, int cols, int relu) {
  int i = (int)(blockIdx.x * blockDim.x + threadIdx.x);
  if (i >= total) return;
  int c = i % cols;
  float x = z[i];
  if (relu) x = fmaxf(x, 0.0f);
  float y = (x - m[c]) * rsqrtf(v[c] + 1e-5f) * g[c] + b[c];
  z[i] = y;
  if (z16) z16[i] = (_Float16)y;
}

// ---------------- final tiny projection ----------------
__global__ void k_final(const float* __restrict__ h2, const float* __restrict__ W,
                        const float* __restrict__ b, float* __restrict__ out,
                        int Bn, int Kd) {
  int i = (int)(blockIdx.x * blockDim.x + threadIdx.x);
  if (i >= Bn) return;
  float s = b[0];
  for (int k = 0; k < Kd; ++k) s += h2[(long)i * Kd + k] * W[k];
  out[i] = s;
}

// ---------------- host orchestration ----------------
extern "C" void kernel_launch(void* const* d_in, const int* in_sizes, int n_in,
                              void* d_out, int out_size, void* d_ws, size_t ws_size,
                              hipStream_t stream) {
  const int B = 64, TOK = 20, D = 512, H = 8, L = 4, S = 512, ODIM = 32;
  const int BC = 8;
  (void)in_sizes; (void)n_in; (void)out_size; (void)ws_size; (void)TOK;

  const float* x      = (const float*)d_in[0];
  const float* in_W   = (const float*)d_in[1];
  const float* in_b   = (const float*)d_in[2];
  const float* qkv_W  = (const float*)d_in[3];
  const float* qkv_b  = (const float*)d_in[4];
  const float* ln1_g  = (const float*)d_in[5];
  const float* ln1_b  = (const float*)d_in[6];
  const float* ffn_W1 = (const float*)d_in[7];
  const float* ffn_b1 = (const float*)d_in[8];
  const float* ffn_W2 = (const float*)d_in[9];
  const float* ffn_b2 = (const float*)d_in[10];
  const float* ln2_g  = (const float*)d_in[11];
  const float* ln2_b  = (const float*)d_in[12];
  const float* out_W  = (const float*)d_in[13];
  const float* out_b  = (const float*)d_in[14];
  const float* d1_W   = (const float*)d_in[15];
  const float* d1_b   = (const float*)d_in[16];
  const float* bn1_g  = (const float*)d_in[17];
  const float* bn1_b  = (const float*)d_in[18];
  const float* bn1_m  = (const float*)d_in[19];
  const float* bn1_v  = (const float*)d_in[20];
  const float* d2_W   = (const float*)d_in[21];
  const float* d2_b   = (const float*)d_in[22];
  const float* bn2_g  = (const float*)d_in[23];
  const float* bn2_b  = (const float*)d_in[24];
  const float* bn2_m  = (const float*)d_in[25];
  const float* bn2_v  = (const float*)d_in[26];
  const float* fin_W  = (const float*)d_in[27];
  const float* fin_b  = (const float*)d_in[28];
  float* outp = (float*)d_out;

  // ---- workspace layout: f32 region, then f16 region, then packed weights ----
  float* fws = (float*)d_ws;
  float* t    = fws;                                // B*S*D   (residual stream)
  float* f    = t    + (long)B * S * D;             // B*S*D   (attn/ffn out)
  float* h1   = f    + (long)B * S * D;             // B*256
  float* h2   = h1   + (long)B * 256;               // B*128
  _Float16* t16    = (_Float16*)(h2 + (long)B * 128);   // B*S*D
  _Float16* qkv16  = t16    + (long)B * S * D;          // BC*S*3D
  _Float16* hid16  = qkv16  + (long)BC * S * 3 * D;     // BC*S*4D
  _Float16* hcat16 = hid16  + (long)BC * S * 4 * D;     // B*2*S*ODIM
  _Float16* h116   = hcat16 + (long)B * 2 * S * ODIM;   // B*256
  _Float16* xpad16 = h116   + (long)B * 256;            // B*2*S*32
  _Float16* pk     = xpad16 + (long)B * 2 * S * 32;     // packed weights

  const long FR = 512;  // f16 elems per fragment
  _Float16* pin  = pk;
  _Float16* pqkv = pin  + 32 * FR;
  _Float16* pff1 = pqkv + (long)4 * 1536 * FR;
  _Float16* pff2 = pff1 + (long)4 * 2048 * FR;
  _Float16* pout = pff2 + (long)4 * 2048 * FR;
  _Float16* pd1  = pout + 32 * FR;
  _Float16* pd2  = pd1  + (long)16384 * FR;

  auto pack = [&](const float* W, _Float16* P, int K, int Kpad, int N, int ldb) {
    int waves = (N / 16) * (Kpad / 32);
    k_pack_b<<<(waves + 7) / 8, 256, 0, stream>>>(W, P, K, Kpad, N, ldb);
  };
  pack(in_W, pin, 20, 32, 512, 512);
  for (int l = 0; l < L; ++l) {
    pack(qkv_W  + (long)l * D * 3 * D, pqkv + (long)l * 1536 * FR, 512, 512, 1536, 1536);
    pack(ffn_W1 + (long)l * D * 4 * D, pff1 + (long)l * 2048 * FR, 512, 512, 2048, 2048);
    pack(ffn_W2 + (long)l * 4 * D * D, pff2 + (long)l * 2048 * FR, 2048, 2048, 512, 512);
  }
  pack(out_W, pout, 512, 512, 32, 32);
  pack(d1_W, pd1, 32768, 32768, 256, 256);
  pack(d2_W, pd2, 256, 256, 128, 128);
  k_pad_x<<<(B * 2 * S * 32 + 255) / 256, 256, 0, stream>>>(x, xpad16, B * 2 * S * 32);

  auto gemm = [&](const _Float16* A, const _Float16* Bp, const float* bias,
                  float* C, _Float16* C16,
                  int M, int N, int K, int lda, int ldc,
                  long sA, long sC, int batch, int relu,
                  int kChunk, int nSplit, int atomicMode) {
    int tiles = (M / 64) * (N / 32);
    dim3 grid((tiles + 7) / 8, batch, nSplit);
    k_gemm<<<grid, 256, 0, stream>>>(A, Bp, bias, C, C16, M, N, K, lda, ldc,
                                     sA, sC, relu, kChunk, atomicMode);
  };

  for (int hf = 0; hf < 2; ++hf) {
    // input projection -> t (f32 residual) + t16 (GEMM input)
    gemm(xpad16 + (long)hf * S * 32, pin, in_b, t, t16,
         S, D, 32, 32, D, (long)2 * S * 32, (long)S * D, B, 0, 32, 1, 0);

    for (int l = 0; l < L; ++l) {
      for (int b0 = 0; b0 < B; b0 += BC) {
        gemm(t16 + (long)b0 * S * D, pqkv + (long)l * 1536 * FR, qkv_b + (long)l * 3 * D,
             nullptr, qkv16, S, 3 * D, D, D, 3 * D,
             (long)S * D, (long)S * 3 * D, BC, 0, D, 1, 0);
        int waves = BC * H * (S / 16);
        k_attn<<<(waves + 3) / 4, 128, 0, stream>>>(qkv16, f + (long)b0 * S * D, waves);
      }
      k_add_ln<<<(B * S + 7) / 8, 256, 0, stream>>>(t, f, ln1_g + (long)l * D,
                                                    ln1_b + (long)l * D, t, t16, B * S);
      for (int b0 = 0; b0 < B; b0 += BC) {
        gemm(t16 + (long)b0 * S * D, pff1 + (long)l * 2048 * FR, ffn_b1 + (long)l * 4 * D,
             nullptr, hid16, S, 4 * D, D, D, 4 * D,
             (long)S * D, (long)S * 4 * D, BC, 1, D, 1, 0);
        gemm(hid16, pff2 + (long)l * 2048 * FR, ffn_b2 + (long)l * D,
             f + (long)b0 * S * D, nullptr, S, D, 4 * D, 4 * D, D,
             (long)S * 4 * D, (long)S * D, BC, 0, 4 * D, 1, 0);
      }
      k_add_ln<<<(B * S + 7) / 8, 256, 0, stream>>>(t, f, ln2_g + (long)l * D,
                                                    ln2_b + (long)l * D, t, t16, B * S);
    }
    gemm(t16, pout, out_b, nullptr, hcat16 + (long)hf * S * ODIM,
         S, ODIM, D, D, ODIM, (long)S * D, (long)2 * S * ODIM, B, 0, D, 1, 0);
  }

  // head: d1 (split-K, f32 atomics) -> bn1(relu)+f16 -> d2 -> bn2 -> fin
  const int K1 = 2 * S * ODIM;            // 32768
  k_row_init<<<(B * 256 + 255) / 256, 256, 0, stream>>>(h1, d1_b, B * 256, 256);
  gemm(hcat16, pd1, nullptr, h1, nullptr, B, 256, K1, K1, 256, 0, 0, 1, 0, 2048, 16, 1);
  k_bn<<<(B * 256 + 255) / 256, 256, 0, stream>>>(h1, h116, bn1_g, bn1_b, bn1_m, bn1_v,
                                                  B * 256, 256, 1);
  gemm(h116, pd2, d2_b, h2, nullptr, B, 128, 256, 256, 128, 0, 0, 1, 1, 256, 1, 0);
  k_bn<<<(B * 128 + 255) / 256, 256, 0, stream>>>(h2, nullptr, bn2_g, bn2_b, bn2_m, bn2_v,
                                                  B * 128, 128, 0);
  k_final<<<1, 64, 0, stream>>>(h2, fin_W, fin_b, outp, B, 128);
}
